// Router_17875653886563
// MI455X (gfx1250) — matrix operand order repaired and checked by reading, not compiled, not run on hardware
//
#include <hip/hip_runtime.h>
#include <hip/hip_bf16.h>

// ---------------------------------------------------------------------------
// MoE router for MI455X (gfx1250, wave32, WMMA).
// logits = X[16384,4096] @ W[64,4096]^T  via v_wmma_f32_16x16x32_bf16,
// then per-token top-2 softmax + aux load-balance loss.
//
// Roofline: X read = 256MB -> 11.5us HBM floor @ 23.3TB/s. To stay on that
// floor we need (a) matrix-core rate (bf16 WMMA, f32 accum), and (b) enough
// waves/SIMD to hide HBM latency. Tiling 16 tokens/wave alone gives only
// 1024 waves (~1/SIMD on a chip this size), so K is split 4-ways: each
// 16-token tile is computed by 4 waves (K=1024 each) whose partial logits
// are combined with ds_add_f32 atomics in LDS -> 4096 waves, 512 blocks,
// identical HBM traffic. B fragments are pre-packed to bf16 in exact
// WMMA-B lane order (512KB, L2-resident) and the inner loop is manually
// double-buffered so WMMAs never wait on same-iteration loads.
// ---------------------------------------------------------------------------

typedef __attribute__((ext_vector_type(16))) __bf16 v16bf;
typedef __attribute__((ext_vector_type(8)))  float  v8f;
typedef __attribute__((ext_vector_type(4)))  float  v4f;

#define NUM_TOKENS   16384
#define EMBED_DIM    4096
#define NUM_EXPERTS  64
#define TOPK         2

#define WAVES_PER_BLOCK   8
#define KSPLIT            4                                     // waves per token tile
#define TILES_PER_BLOCK   (WAVES_PER_BLOCK / KSPLIT)            // 2
#define TOKENS_PER_WAVE   16
#define TOKENS_PER_BLOCK  (TILES_PER_BLOCK * TOKENS_PER_WAVE)   // 32
#define K_STEP            32
#define K_PER_WAVE        (EMBED_DIM / KSPLIT)                  // 1024
#define NFRAG             (NUM_EXPERTS / 16)                    // 4
#define WSTEP_ELEMS       (NFRAG * 32 * 16)                     // 2048 bf16 per K-step

// ---------------------------------------------------------------------------
// Kernel 1: pack W (f32 [64,4096]) into bf16 in WMMA-B fragment lane order.
// Flat packed index = (((k0/32)*4 + frag)*32 + lane)*16 + i, where
//   expert = frag*16 + (lane&15),  k = k0 + (lane>>4)*8 + (i&7) + (i>>3)*16
// (ISA 16-bit B 32x16: lanes 0-15 hold K=0..7,16..23, lanes 16-31 hold K=8..15,24..31).
// Also zero-inits the stats region (P_sum[64], counts[64]).
// ---------------------------------------------------------------------------
__global__ void prepack_w_kernel(const float* __restrict__ W,
                                 unsigned short* __restrict__ Wb,
                                 float* __restrict__ stats) {
    int idx = blockIdx.x * blockDim.x + threadIdx.x;     // 0 .. 262143
    int i     =  idx        & 15;
    int lane  = (idx >> 4)  & 31;
    int nfrag = (idx >> 9)  &  3;
    int kstep =  idx >> 11;
    int expert = nfrag * 16 + (lane & 15);
    int k      = kstep * K_STEP + ((lane >> 4) * 8) + (i & 7) + ((i >> 3) * 16);
    __bf16 h = (__bf16)W[expert * EMBED_DIM + k];
    Wb[idx] = *(unsigned short*)&h;
    if (blockIdx.x == 0 && threadIdx.x < 2 * NUM_EXPERTS)
        stats[threadIdx.x] = 0.0f;
}

// ---------------------------------------------------------------------------
// Kernel 2: WMMA GEMM (K-split x4) + top-2 + softmax + stats accumulation.
// grid = 512 blocks x 256 threads (8 waves). Wave w: token tile w>>2,
// K-quarter w&3. Partial logits combined via ds_add_f32 in LDS.
// ---------------------------------------------------------------------------
__global__ void __launch_bounds__(256)
router_gemm_kernel(const float* __restrict__ X,
                   const unsigned short* __restrict__ Wb_u16,
                   float* __restrict__ out_w,     // [16384,2] routing weights
                   float* __restrict__ out_e,     // [16384,2] expert ids (as float)
                   float* __restrict__ g_P,       // [64] sum of softmax probs
                   float* __restrict__ g_cnt) {   // [64] selection counts
    __shared__ float    lds_logits[TOKENS_PER_BLOCK * NUM_EXPERTS]; // 8KB
    __shared__ float    s_max[TOKENS_PER_BLOCK];
    __shared__ float    s_zinv[TOKENS_PER_BLOCK];
    __shared__ float    s_P[NUM_EXPERTS];
    __shared__ unsigned s_cnt[NUM_EXPERTS];

    const int wave = threadIdx.x >> 5;
    const int lane = threadIdx.x & 31;
    const int hi   = lane >> 4;            // 0/1: lane half selects K sub-group
    const int col  = lane & 15;
    const int tileIdx = wave >> 2;         // 0..1: token tile within block
    const int kq      = wave & (KSPLIT - 1);

    // zero the logits accumulation tile + stats scratch
    for (int i = threadIdx.x; i < TOKENS_PER_BLOCK * NUM_EXPERTS; i += 256)
        lds_logits[i] = 0.0f;
    if (threadIdx.x < NUM_EXPERTS) { s_P[threadIdx.x] = 0.0f; s_cnt[threadIdx.x] = 0u; }
    __syncthreads();

    const int token_base = (blockIdx.x * TILES_PER_BLOCK + tileIdx) * TOKENS_PER_WAVE;
    const int kbase = kq * K_PER_WAVE;
    const int kend  = kbase + K_PER_WAVE;

    const float*  xrow = X + (size_t)(token_base + col) * EMBED_DIM;
    const __bf16* Wb   = (const __bf16*)Wb_u16;

    v8f acc0 = {}, acc1 = {}, acc2 = {}, acc3 = {};

    // ---- double-buffered main loop -------------------------------------
    // A fragment (16-bit A 16x32 layout):
    //   lanes 0-15 : elems 0..7 = K 0..7,  elems 8..15 = K 16..23
    //   lanes 16-31: elems 0..7 = K 8..15, elems 8..15 = K 24..31
    const float*  xp   = xrow + kbase + hi * 8;
    const __bf16* wcur = Wb + (size_t)(kbase >> 5) * WSTEP_ELEMS + (size_t)lane * 16;

    v4f a0 = *(const v4f*)(xp);
    v4f a1 = *(const v4f*)(xp + 4);
    v4f a2 = *(const v4f*)(xp + 16);
    v4f a3 = *(const v4f*)(xp + 20);
    v16bf b0 = *(const v16bf*)(wcur + 0 * 512);
    v16bf b1 = *(const v16bf*)(wcur + 1 * 512);
    v16bf b2 = *(const v16bf*)(wcur + 2 * 512);
    v16bf b3 = *(const v16bf*)(wcur + 3 * 512);

    for (int k0 = kbase; k0 < kend; k0 += K_STEP) {
        const bool    more = (k0 + K_STEP) < kend;
        const __bf16* wnxt = more ? (wcur + WSTEP_ELEMS) : wcur;   // clamp last iter
        const float*  xnxt = more ? (xp + K_STEP)        : xp;

        // convert current A to bf16 (packed cvts)
        v16bf a;
#pragma unroll
        for (int i = 0; i < 4; i++) a[i]      = (__bf16)a0[i];
#pragma unroll
        for (int i = 0; i < 4; i++) a[4 + i]  = (__bf16)a1[i];
#pragma unroll
        for (int i = 0; i < 4; i++) a[8 + i]  = (__bf16)a2[i];
#pragma unroll
        for (int i = 0; i < 4; i++) a[12 + i] = (__bf16)a3[i];

        // issue next iteration's loads before this iteration's WMMAs
        v4f   na0 = *(const v4f*)(xnxt);
        v4f   na1 = *(const v4f*)(xnxt + 4);
        v4f   na2 = *(const v4f*)(xnxt + 16);
        v4f   na3 = *(const v4f*)(xnxt + 20);
        v16bf nb0 = *(const v16bf*)(wnxt + 0 * 512);
        v16bf nb1 = *(const v16bf*)(wnxt + 1 * 512);
        v16bf nb2 = *(const v16bf*)(wnxt + 2 * 512);
        v16bf nb3 = *(const v16bf*)(wnxt + 3 * 512);
        __builtin_prefetch(xnxt + 2 * K_STEP, 0, 1);   // global_prefetch_b8

        acc0 = __builtin_amdgcn_wmma_f32_16x16x32_bf16(false, a, false, b0, (short)0, acc0, false, false);
        acc1 = __builtin_amdgcn_wmma_f32_16x16x32_bf16(false, a, false, b1, (short)0, acc1, false, false);
        acc2 = __builtin_amdgcn_wmma_f32_16x16x32_bf16(false, a, false, b2, (short)0, acc2, false, false);
        acc3 = __builtin_amdgcn_wmma_f32_16x16x32_bf16(false, a, false, b3, (short)0, acc3, false, false);

        a0 = na0; a1 = na1; a2 = na2; a3 = na3;
        b0 = nb0; b1 = nb1; b2 = nb2; b3 = nb3;
        wcur = wnxt; xp = xnxt;
    }

    // ---- combine K-split partials into LDS (ds_add_f32).
    // C layout: VGPR r, lanes 0-15 -> (M=r, N=lane), lanes 16-31 -> (M=8+r).
    float* tile = lds_logits + (tileIdx * TOKENS_PER_WAVE) * NUM_EXPERTS;
#pragma unroll
    for (int r = 0; r < 8; r++) {
        int tok = r + 8 * hi;
        atomicAdd(&tile[tok * NUM_EXPERTS +  0 + col], acc0[r]);
        atomicAdd(&tile[tok * NUM_EXPERTS + 16 + col], acc1[r]);
        atomicAdd(&tile[tok * NUM_EXPERTS + 32 + col], acc2[r]);
        atomicAdd(&tile[tok * NUM_EXPERTS + 48 + col], acc3[r]);
    }
    __syncthreads();

    // ---- per-token: top-2, softmax normalizer, outputs, count histogram.
    if (threadIdx.x < TOKENS_PER_BLOCK) {
        const float* row = lds_logits + threadIdx.x * NUM_EXPERTS;
        float m1 = -3.0e38f, m2 = -3.0e38f;
        int   i1 = 0, i2 = 0;
#pragma unroll 4
        for (int e = 0; e < NUM_EXPERTS; e++) {
            float v = row[e];
            if (v > m1)      { m2 = m1; i2 = i1; m1 = v; i1 = e; }
            else if (v > m2) { m2 = v; i2 = e; }
        }
        float z = 0.0f;
#pragma unroll 4
        for (int e = 0; e < NUM_EXPERTS; e++) z += __expf(row[e] - m1);
        s_max[threadIdx.x]  = m1;
        s_zinv[threadIdx.x] = 1.0f / z;

        // softmax over the two selected logits
        float e2  = __expf(m2 - m1);
        float inv = 1.0f / (1.0f + e2);
        int gt = blockIdx.x * TOKENS_PER_BLOCK + threadIdx.x;
        out_w[gt * TOPK + 0] = inv;
        out_w[gt * TOPK + 1] = e2 * inv;
        out_e[gt * TOPK + 0] = (float)i1;
        out_e[gt * TOPK + 1] = (float)i2;
        atomicAdd(&s_cnt[i1], 1u);
        atomicAdd(&s_cnt[i2], 1u);
    }
    __syncthreads();

    // ---- P accumulation: thread (g,e) sums probs of expert e, tokens g::4.
    {
        int e = threadIdx.x & (NUM_EXPERTS - 1);
        int g = threadIdx.x >> 6;                 // 0..3
        float acc = 0.0f;
        for (int t = g; t < TOKENS_PER_BLOCK; t += 4)
            acc += __expf(lds_logits[t * NUM_EXPERTS + e] - s_max[t]) * s_zinv[t];
        atomicAdd(&s_P[e], acc);
    }
    __syncthreads();

    if (threadIdx.x < NUM_EXPERTS) {
        atomicAdd(&g_P[threadIdx.x],   s_P[threadIdx.x]);
        atomicAdd(&g_cnt[threadIdx.x], (float)s_cnt[threadIdx.x]);
    }
}

// ---------------------------------------------------------------------------
// Kernel 3: aux_loss = E * sum_e( f_e * P_e ),
//   f_e = counts_e / (T*K),  P_e = P_sum_e / T
// ---------------------------------------------------------------------------
__global__ void finalize_kernel(const float* __restrict__ stats,
                                float* __restrict__ out_aux) {
    __shared__ float red[NUM_EXPERTS];
    int e = threadIdx.x;
    float f  = stats[NUM_EXPERTS + e] * (1.0f / (float)(NUM_TOKENS * TOPK));
    float Pm = stats[e]               * (1.0f / (float)NUM_TOKENS);
    red[e] = f * Pm;
    __syncthreads();
    for (int s = NUM_EXPERTS / 2; s > 0; s >>= 1) {
        if (e < s) red[e] += red[e + s];
        __syncthreads();
    }
    if (e == 0) *out_aux = (float)NUM_EXPERTS * red[0];
}

// ---------------------------------------------------------------------------
extern "C" void kernel_launch(void* const* d_in, const int* in_sizes, int n_in,
                              void* d_out, int out_size, void* d_ws, size_t ws_size,
                              hipStream_t stream) {
    const float* X = (const float*)d_in[0];   // [16384, 4096]
    const float* W = (const float*)d_in[1];   // [64, 4096]
    float* out = (float*)d_out;
    float* out_w   = out;                     // [16384,2]
    float* out_e   = out + TOPK * NUM_TOKENS; // [16384,2] expert ids as float
    float* out_aux = out + 2 * TOPK * NUM_TOKENS;

    // workspace: packed bf16 W (512KB) then stats[128] (P_sum[64], counts[64])
    unsigned short* Wb = (unsigned short*)d_ws;
    float* stats = (float*)((char*)d_ws + (size_t)NUM_EXPERTS * EMBED_DIM * 2);
    float* g_P   = stats;
    float* g_cnt = stats + NUM_EXPERTS;

    prepack_w_kernel<<<(NUM_EXPERTS * EMBED_DIM) / 256, 256, 0, stream>>>(W, Wb, stats);
    router_gemm_kernel<<<NUM_TOKENS / TOKENS_PER_BLOCK, 256, 0, stream>>>(
        X, Wb, out_w, out_e, g_P, g_cnt);
    finalize_kernel<<<1, NUM_EXPERTS, 0, stream>>>(stats, out_aux);
}
